// ResEGNN_79783312490626
// MI455X (gfx1250) — matrix operand order repaired and checked by reading, not compiled, not run on hardware
//
#include <hip/hip_runtime.h>
#include <hip/hip_bf16.h>

#define NN   50000
#define NE   800000
#define HID  64
#define INF  21
#define OUTF 20
#define NL   4

typedef __attribute__((ext_vector_type(16))) __bf16 v16bf;
typedef __attribute__((ext_vector_type(8)))  float  v8f;

struct __align__(16) U4 { unsigned x, y, z, w; };

static __device__ __forceinline__ unsigned short f2bf(float f) {
    unsigned u = __float_as_uint(f);
    u += 0x7FFFu + ((u >> 16) & 1u);          // round-to-nearest-even
    return (unsigned short)(u >> 16);
}
static __device__ __forceinline__ unsigned pack2bf(float a, float b) {
    return (unsigned)f2bf(a) | ((unsigned)f2bf(b) << 16);
}
static __device__ __forceinline__ float siluf(float x) {
    return x / (1.0f + __expf(-x));
}
static __device__ __forceinline__ v16bf frag_pair(U4 a, U4 b) {
    union { U4 q[2]; v16bf v; } u;
    u.q[0] = a; u.q[1] = b;
    return u.v;
}
static __device__ __forceinline__ v8f vzero() {
    v8f z;
#pragma unroll
    for (int i = 0; i < 8; ++i) z[i] = 0.0f;
    return z;
}
static __device__ __forceinline__ v8f wmma_bf16(v16bf a, v16bf b, v8f c) {
    // D(16x16,f32) = A(16x32,bf16) * B(32x16,bf16) + C
    return __builtin_amdgcn_wmma_f32_16x16x32_bf16(false, a, false, b,
                                                   (short)0, c, false, false);
}

// ---------------------------------------------------------------------------
// Weight fragment pre-swizzle: row-major W[K x 64] (fp32) -> bf16 B-fragments.
// Fragment f = kc*4 + nt covers K in [32kc,32kc+32), N in [16nt,16nt+16).
// CDNA5 16-bit B layout: lane l holds N=l%16, K=(v>>2)*16+(l<16?0:8)+2*(v&3)+h in dword v.
// Stored lane-contiguous: out[(f*32+lane)*8 + v]  (32B per lane per fragment).
// ---------------------------------------------------------------------------
__global__ void k_prep_frags(const float* __restrict__ W,
                             unsigned* __restrict__ out, int kchunks) {
    int tid = blockIdx.x * blockDim.x + threadIdx.x;
    int total = kchunks * 4 * 32 * 8;
    if (tid >= total) return;
    int v    = tid & 7;
    int lane = (tid >> 3) & 31;
    int nt   = (tid >> 8) & 3;
    int kc   = tid >> 10;
    int half = lane >> 4;
    int n    = lane & 15;
    int klo  = kc * 32 + (v >> 2) * 16 + half * 8 + 2 * (v & 3);
    int N    = nt * 16 + n;
    out[(size_t)((kc * 4 + nt) * 32 + lane) * 8 + v] =
        pack2bf(W[(size_t)klo * HID + N], W[(size_t)(klo + 1) * HID + N]);
}

// h = h_in @ emb_in_w + emb_in_b ; also emit bf16 copy
__global__ void k_embed(const float* __restrict__ hin, const float* __restrict__ W,
                        const float* __restrict__ b, float* __restrict__ h,
                        unsigned short* __restrict__ hbf) {
    int i = blockIdx.x * blockDim.x + threadIdx.x;
    if (i >= NN * HID) return;
    int node = i >> 6, j = i & 63;
    float s = b[j];
#pragma unroll
    for (int k = 0; k < INF; ++k) s += hin[node * INF + k] * W[k * HID + j];
    h[i] = s;
    hbf[i] = f2bf(s);
}

__global__ void k_count(const int* __restrict__ row, float* __restrict__ cnt) {
    int i = blockIdx.x * blockDim.x + threadIdx.x;
    if (i < NE) atomicAdd(&cnt[row[i]], 1.0f);
}

__global__ void k_coord_update(float* __restrict__ x, const float* __restrict__ xagg,
                               const float* __restrict__ cnt) {
    int i = blockIdx.x * blockDim.x + threadIdx.x;
    if (i >= NN * 3) return;
    x[i] += xagg[i] / fmaxf(cnt[i / 3], 1.0f);
}

__global__ void k_aggbf(const float* __restrict__ agg, unsigned short* __restrict__ aggbf) {
    int i = blockIdx.x * blockDim.x + threadIdx.x;
    if (i < NN * HID) aggbf[i] = f2bf(agg[i]);
}

// ---------------------------------------------------------------------------
// Edge kernel: 4 waves/block, 16 edges/wave (12500 blocks * 64 = 800000 exact)
// Weight fragments (W1|W2|CW1 = 8192 dwords) staged once per block in LDS;
// B operands then come from ds_load_b128 (short dscnt waits instead of full
// loadcnt drains before every WMMA).
// GEMM1: [h_i,h_j,r](16x129) @ W1 -> silu  (radial col = rank-1 VALU update)
// GEMM2: @ W2 -> silu = m ; atomic agg[row] += m
// GEMM3: m @ CW1 -> silu ; s = . @ cw2 ; atomic xagg[row] += diff*s
// ---------------------------------------------------------------------------
__global__ __launch_bounds__(128) void k_edge(
    const int* __restrict__ row, const int* __restrict__ col,
    const float* __restrict__ x, const unsigned short* __restrict__ hbf,
    const unsigned* __restrict__ fAll,   // W1(16 frags)|W2(8)|CW1(8), contiguous
    const float* __restrict__ w1last, const float* __restrict__ eb1,
    const float* __restrict__ eb2, const float* __restrict__ cb1,
    const float* __restrict__ cw2,
    float* __restrict__ agg, float* __restrict__ xagg) {
    __shared__ __align__(16) unsigned wlds[8192];          // 32 KB weight frags
    __shared__ __align__(16) unsigned short mbuf[4][16 * HID];
    __shared__ float sbuf[4][16];

    const int tid  = threadIdx.x;
    const int wave = tid >> 5, lane = tid & 31;
    const int half = lane >> 4, n = lane & 15;
    const int tb   = (blockIdx.x * 4 + wave) * 16;
    const int e    = tb + n;
    const int r0 = row[e], c0 = col[e];
    const float dx = x[r0 * 3 + 0] - x[c0 * 3 + 0];
    const float dy = x[r0 * 3 + 1] - x[c0 * 3 + 1];
    const float dz = x[r0 * 3 + 2] - x[c0 * 3 + 2];
    const float radial = dx * dx + dy * dy + dz * dz;

    // cooperative weight staging: 2048 x 16B, 16 b128 loads per thread
    {
        const U4* src = (const U4*)fAll;
        U4* dst = (U4*)wlds;
#pragma unroll
        for (int i = 0; i < 16; ++i) dst[tid + i * 128] = src[tid + i * 128];
    }
    __syncthreads();

    // LDS B-fragment fetch (frag id f: W1=0..15, W2=16..23, CW1=24..31)
    auto bfrag = [&](int f) {
        const U4* bp = (const U4*)(wlds + (size_t)(f * 32 + lane) * 8);
        return frag_pair(bp[0], bp[1]);
    };

    // ---- GEMM1 ----
    v8f acc[4];
#pragma unroll
    for (int t = 0; t < 4; ++t) acc[t] = vzero();
    const U4* pr = (const U4*)(hbf + (size_t)r0 * HID);
    const U4* pc = (const U4*)(hbf + (size_t)c0 * HID);
#pragma unroll
    for (int c = 0; c < 4; ++c) {
        const U4* p = (c < 2) ? pr : pc;
        const int k8 = (c & 1) ? 4 : 0;                 // kb/8 (16B units)
        v16bf a = frag_pair(p[k8 + half], p[k8 + 2 + half]);
#pragma unroll
        for (int t = 0; t < 4; ++t)
            acc[t] = wmma_bf16(a, bfrag(c * 4 + t), acc[t]);
    }
    // radial rank-1 + bias + silu
#pragma unroll
    for (int t = 0; t < 4; ++t) {
        const float wl = w1last[t * 16 + n];
        const float bb = eb1[t * 16 + n];
#pragma unroll
        for (int v = 0; v < 8; ++v) {
            float rv = __shfl(radial, v + half * 8, 32);
            acc[t][v] = siluf(acc[t][v] + rv * wl + bb);
        }
    }
#pragma unroll
    for (int t = 0; t < 4; ++t)
#pragma unroll
        for (int v = 0; v < 8; ++v)
            mbuf[wave][(v + half * 8) * HID + t * 16 + n] = f2bf(acc[t][v]);
    __syncthreads();

    // ---- GEMM2 ----
    v8f acc2[4];
#pragma unroll
    for (int t = 0; t < 4; ++t) acc2[t] = vzero();
    const U4* pm = (const U4*)(&mbuf[wave][n * HID]);
#pragma unroll
    for (int c = 0; c < 2; ++c) {
        v16bf a = frag_pair(pm[c * 4 + half], pm[c * 4 + 2 + half]);
#pragma unroll
        for (int t = 0; t < 4; ++t)
            acc2[t] = wmma_bf16(a, bfrag(16 + c * 4 + t), acc2[t]);
    }
#pragma unroll
    for (int t = 0; t < 4; ++t) {
        const float bb = eb2[t * 16 + n];
#pragma unroll
        for (int v = 0; v < 8; ++v) acc2[t][v] = siluf(acc2[t][v] + bb);
    }
    // agg[row_e] += m   (M = v + half*8 ; row of that edge via shuffle)
#pragma unroll
    for (int t = 0; t < 4; ++t)
#pragma unroll
        for (int v = 0; v < 8; ++v) {
            int rM = __shfl(r0, v + half * 8, 32);
            atomicAdd(&agg[(size_t)rM * HID + t * 16 + n], acc2[t][v]);
        }
    __syncthreads();
#pragma unroll
    for (int t = 0; t < 4; ++t)
#pragma unroll
        for (int v = 0; v < 8; ++v)
            mbuf[wave][(v + half * 8) * HID + t * 16 + n] = f2bf(acc2[t][v]);
    __syncthreads();

    // ---- GEMM3 + coord head ----
    v8f acc3[4];
#pragma unroll
    for (int t = 0; t < 4; ++t) acc3[t] = vzero();
#pragma unroll
    for (int c = 0; c < 2; ++c) {
        v16bf a = frag_pair(pm[c * 4 + half], pm[c * 4 + 2 + half]);
#pragma unroll
        for (int t = 0; t < 4; ++t)
            acc3[t] = wmma_bf16(a, bfrag(24 + c * 4 + t), acc3[t]);
    }
    float s[8];
#pragma unroll
    for (int v = 0; v < 8; ++v) s[v] = 0.0f;
#pragma unroll
    for (int t = 0; t < 4; ++t) {
        const float cb = cb1[t * 16 + n];
        const float wv = cw2[t * 16 + n];
#pragma unroll
        for (int v = 0; v < 8; ++v) s[v] += siluf(acc3[t][v] + cb) * wv;
    }
#pragma unroll
    for (int v = 0; v < 8; ++v) {
        s[v] += __shfl_xor(s[v], 1, 32);
        s[v] += __shfl_xor(s[v], 2, 32);
        s[v] += __shfl_xor(s[v], 4, 32);
        s[v] += __shfl_xor(s[v], 8, 32);
    }
    if (n == 0) {
#pragma unroll
        for (int v = 0; v < 8; ++v) sbuf[wave][v + half * 8] = s[v];
    }
    __syncthreads();
    if (lane < 16) {
        float sc = sbuf[wave][lane];
        atomicAdd(&xagg[(size_t)r0 * 3 + 0], dx * sc);
        atomicAdd(&xagg[(size_t)r0 * 3 + 1], dy * sc);
        atomicAdd(&xagg[(size_t)r0 * 3 + 2], dz * sc);
    }
}

// ---------------------------------------------------------------------------
// Node kernel: h += silu([h|agg] @ W1 + b1) @ W2 + b2 ; 16 nodes/wave.
// NW1|NW2 fragments (6144 dwords = 24 KB) staged in LDS per block.
// ---------------------------------------------------------------------------
__global__ __launch_bounds__(128) void k_node(
    unsigned short* __restrict__ hbf, float* __restrict__ h,
    const unsigned short* __restrict__ aggbf,
    const unsigned* __restrict__ fAll,   // NW1(16 frags)|NW2(8), contiguous
    const float* __restrict__ nb1, const float* __restrict__ nb2) {
    __shared__ __align__(16) unsigned wlds[6144];          // 24 KB weight frags
    __shared__ __align__(16) unsigned short mbuf[4][16 * HID];

    const int tid  = threadIdx.x;
    const int wave = tid >> 5, lane = tid & 31;
    const int half = lane >> 4, n = lane & 15;
    const int tb   = (blockIdx.x * 4 + wave) * 16;
    int nodeA = tb + n;
    if (nodeA >= NN) nodeA = NN - 1;                    // keep EXEC full; clamp loads

    {
        const U4* src = (const U4*)fAll;
        U4* dst = (U4*)wlds;
#pragma unroll
        for (int i = 0; i < 12; ++i) dst[tid + i * 128] = src[tid + i * 128];
    }
    __syncthreads();

    auto bfrag = [&](int f) {
        const U4* bp = (const U4*)(wlds + (size_t)(f * 32 + lane) * 8);
        return frag_pair(bp[0], bp[1]);
    };

    v8f acc[4];
#pragma unroll
    for (int t = 0; t < 4; ++t) acc[t] = vzero();
    const U4* ph = (const U4*)(hbf + (size_t)nodeA * HID);
    const U4* pa = (const U4*)(aggbf + (size_t)nodeA * HID);
#pragma unroll
    for (int c = 0; c < 4; ++c) {
        const U4* p = (c < 2) ? ph : pa;
        const int k8 = (c & 1) ? 4 : 0;
        v16bf a = frag_pair(p[k8 + half], p[k8 + 2 + half]);
#pragma unroll
        for (int t = 0; t < 4; ++t)
            acc[t] = wmma_bf16(a, bfrag(c * 4 + t), acc[t]);
    }
#pragma unroll
    for (int t = 0; t < 4; ++t) {
        const float bb = nb1[t * 16 + n];
#pragma unroll
        for (int v = 0; v < 8; ++v) acc[t][v] = siluf(acc[t][v] + bb);
    }
#pragma unroll
    for (int t = 0; t < 4; ++t)
#pragma unroll
        for (int v = 0; v < 8; ++v)
            mbuf[wave][(v + half * 8) * HID + t * 16 + n] = f2bf(acc[t][v]);
    __syncthreads();

    v8f acc2[4];
#pragma unroll
    for (int t = 0; t < 4; ++t) acc2[t] = vzero();
    const U4* pm = (const U4*)(&mbuf[wave][n * HID]);
#pragma unroll
    for (int c = 0; c < 2; ++c) {
        v16bf a = frag_pair(pm[c * 4 + half], pm[c * 4 + 2 + half]);
#pragma unroll
        for (int t = 0; t < 4; ++t)
            acc2[t] = wmma_bf16(a, bfrag(16 + c * 4 + t), acc2[t]);
    }
#pragma unroll
    for (int t = 0; t < 4; ++t)
#pragma unroll
        for (int v = 0; v < 8; ++v) {
            int node = tb + v + half * 8;
            if (node < NN) {
                size_t idx = (size_t)node * HID + t * 16 + n;
                float nh = h[idx] + acc2[t][v] + nb2[t * 16 + n];
                h[idx] = nh;
                hbf[idx] = f2bf(nh);
            }
        }
}

// Y[N,64] = X[N,64] @ W[64,64] + b   (optional relu)
__global__ void k_gemm64(const float* __restrict__ X, const float* __restrict__ W,
                         const float* __restrict__ b, float* __restrict__ Y,
                         int dorelu) {
    int i = blockIdx.x * blockDim.x + threadIdx.x;
    if (i >= NN * HID) return;
    int node = i >> 6, j = i & 63;
    float s = b[j];
#pragma unroll 8
    for (int k = 0; k < HID; ++k) s += X[(size_t)node * HID + k] * W[k * HID + j];
    Y[i] = dorelu ? fmaxf(s, 0.0f) : s;
}

__global__ void k_head_out(const float* __restrict__ X, const float* __restrict__ W,
                           const float* __restrict__ b, float* __restrict__ out) {
    int i = blockIdx.x * blockDim.x + threadIdx.x;
    if (i >= NN * OUTF) return;
    int node = i / OUTF, j = i % OUTF;
    float s = b[j];
#pragma unroll 8
    for (int k = 0; k < HID; ++k) s += X[(size_t)node * HID + k] * W[k * OUTF + j];
    out[i] = s;
}

// ---------------------------------------------------------------------------
extern "C" void kernel_launch(void* const* d_in, const int* in_sizes, int n_in,
                              void* d_out, int out_size, void* d_ws, size_t ws_size,
                              hipStream_t stream) {
    (void)in_sizes; (void)n_in; (void)out_size; (void)ws_size;
    const float* h_in      = (const float*)d_in[0];
    const float* x_in      = (const float*)d_in[1];
    const int*   edges     = (const int*)d_in[2];
    const float* emb_in_w  = (const float*)d_in[3];
    const float* emb_in_b  = (const float*)d_in[4];
    const float* emb_out_w = (const float*)d_in[5];
    const float* emb_out_b = (const float*)d_in[6];
    const float* edge_w1   = (const float*)d_in[7];
    const float* edge_b1   = (const float*)d_in[8];
    const float* edge_w2   = (const float*)d_in[9];
    const float* edge_b2   = (const float*)d_in[10];
    const float* node_w1   = (const float*)d_in[11];
    const float* node_b1   = (const float*)d_in[12];
    const float* node_w2   = (const float*)d_in[13];
    const float* node_b2   = (const float*)d_in[14];
    const float* coord_w1  = (const float*)d_in[15];
    const float* coord_b1  = (const float*)d_in[16];
    const float* coord_w2  = (const float*)d_in[17];
    const float* head_w1   = (const float*)d_in[18];
    const float* head_b1   = (const float*)d_in[19];
    const float* head_w2   = (const float*)d_in[20];
    const float* head_b2   = (const float*)d_in[21];
    float* out = (float*)d_out;

    const int* row = edges;
    const int* col = edges + NE;

    // workspace carve (256B aligned)
    char* base = (char*)d_ws;
    size_t off = 0;
    auto carve = [&](size_t bytes) -> char* {
        char* p = base + off;
        off += (bytes + 255) & ~(size_t)255;
        return p;
    };
    float*          h     = (float*)carve((size_t)NN * HID * 4);
    unsigned short* hbf   = (unsigned short*)carve((size_t)NN * HID * 2);
    float*          xcur  = (float*)carve((size_t)NN * 3 * 4);
    float*          cnt   = (float*)carve((size_t)NN * 4);
    float*          agg   = (float*)carve((size_t)NN * HID * 4);
    unsigned short* aggbf = (unsigned short*)carve((size_t)NN * HID * 2);
    float*          xagg  = (float*)carve((size_t)NN * 3 * 4);
    float*          t2    = (float*)carve((size_t)NN * HID * 4);
    unsigned*       frags = (unsigned*)carve((size_t)NL * 14336 * 4);
    float*          t1    = agg;   // agg is dead after the last layer

    const size_t LSTRIDE = 14336;  // dwords per layer of fragments
    auto fEDGE = [&](int l) { return frags + l * LSTRIDE; };          // W1|W2|CW1 (8192 dw)
    auto fEW2  = [&](int l) { return frags + l * LSTRIDE + 4096; };
    auto fCW1  = [&](int l) { return frags + l * LSTRIDE + 6144; };
    auto fNODE = [&](int l) { return frags + l * LSTRIDE + 8192; };   // NW1|NW2 (6144 dw)
    auto fNW2  = [&](int l) { return frags + l * LSTRIDE + 12288; };

    // setup
    hipMemsetAsync(cnt, 0, (size_t)NN * 4, stream);
    hipMemcpyAsync(xcur, x_in, (size_t)NN * 3 * 4, hipMemcpyDeviceToDevice, stream);
    k_embed<<<(NN * HID + 255) / 256, 256, 0, stream>>>(h_in, emb_in_w, emb_in_b, h, hbf);
    k_count<<<(NE + 255) / 256, 256, 0, stream>>>(row, cnt);
    for (int l = 0; l < NL; ++l) {
        k_prep_frags<<<32, 128, 0, stream>>>(edge_w1 + (size_t)l * 129 * HID, fEDGE(l), 4);
        k_prep_frags<<<16, 128, 0, stream>>>(edge_w2 + (size_t)l * HID * HID, fEW2(l), 2);
        k_prep_frags<<<16, 128, 0, stream>>>(coord_w1 + (size_t)l * HID * HID, fCW1(l), 2);
        k_prep_frags<<<32, 128, 0, stream>>>(node_w1 + (size_t)l * 128 * HID, fNODE(l), 4);
        k_prep_frags<<<16, 128, 0, stream>>>(node_w2 + (size_t)l * HID * HID, fNW2(l), 2);
    }

    // layers
    const int EDGE_BLOCKS = NE / 64;                 // 12500 (exact)
    const int NODE_BLOCKS = (NN + 63) / 64;          // 782
    for (int l = 0; l < NL; ++l) {
        hipMemsetAsync(agg, 0, (size_t)NN * HID * 4, stream);
        hipMemsetAsync(xagg, 0, (size_t)NN * 3 * 4, stream);
        k_edge<<<EDGE_BLOCKS, 128, 0, stream>>>(
            row, col, xcur, hbf, fEDGE(l),
            edge_w1 + (size_t)l * 129 * HID + 128 * HID,
            edge_b1 + l * HID, edge_b2 + l * HID,
            coord_b1 + l * HID, coord_w2 + l * HID, agg, xagg);
        k_coord_update<<<(NN * 3 + 255) / 256, 256, 0, stream>>>(xcur, xagg, cnt);
        k_aggbf<<<(NN * HID + 255) / 256, 256, 0, stream>>>(agg, aggbf);
        k_node<<<NODE_BLOCKS, 128, 0, stream>>>(hbf, h, aggbf, fNODE(l),
                                                node_b1 + l * HID, node_b2 + l * HID);
    }

    // head: h@emb_out+b -> relu(.@head_w1+b1) -> .@head_w2+b2
    k_gemm64<<<(NN * HID + 255) / 256, 256, 0, stream>>>(h, emb_out_w, emb_out_b, t1, 0);
    k_gemm64<<<(NN * HID + 255) / 256, 256, 0, stream>>>(t1, head_w1, head_b1, t2, 1);
    k_head_out<<<(NN * OUTF + 255) / 256, 256, 0, stream>>>(t2, head_w2, head_b2, out);
}
